// Encoder_74285754351733
// MI455X (gfx1250) — matrix operand order repaired and checked by reading, not compiled, not run on hardware
//
#include <hip/hip_runtime.h>
#include <hip/hip_bf16.h>
#include <stddef.h>

// ---------------------------------------------------------------------------
// Types for CDNA5 WMMA (wave32): v_wmma_f32_16x16x32_bf16
// ---------------------------------------------------------------------------
typedef __attribute__((ext_vector_type(16))) __bf16 v16bf;
typedef __attribute__((ext_vector_type(8)))  float  v8f;
typedef __attribute__((ext_vector_type(4)))  unsigned int v4u;

#define BB   128
#define TT   256
#define HH   1024
#define G4H  4096    // 4*H
#define NBLK 64      // blocks in persistent scan kernel (one per hidden tile)

__device__ __forceinline__ v8f wmma_bf16(v16bf a, v16bf b, v8f c) {
  return __builtin_amdgcn_wmma_f32_16x16x32_bf16(
      false, a, false, b, (short)0, c, false, false);
}

// ---------------------------------------------------------------------------
// WMMA fragment loaders (bf16, 16x16x32 shape).
//
// A (16Mx32K, row-major, ld = K-stride):
//   lane L holds row M = L&15; kbase = (L>>4)*16
//   halves j=0..15 -> K = kbase+j
//
// B (32Kx16N) from pre-transposed BT[N][K] (row-major, ld=K):
//   lane L holds column N = L&15; halves j=0..15 -> K = (L>>4)*16 + j
// ---------------------------------------------------------------------------
__device__ __forceinline__ v16bf load_a_frag(const __bf16* __restrict__ A, int lda) {
  int lane = threadIdx.x & 31;
  int row  = lane & 15;
  int kb   = (lane >> 4) * 16;
  const __bf16* p = A + (size_t)row * lda + kb;
  union { v4u u[2]; v16bf v; } u;
  u.u[0] = *(const v4u*)(p);
  u.u[1] = *(const v4u*)(p + 8);
  return u.v;
}

__device__ __forceinline__ v16bf load_b_frag(const __bf16* __restrict__ BT, int ldb) {
  int lane = threadIdx.x & 31;
  int col  = lane & 15;
  int kb   = (lane >> 4) * 16;
  const __bf16* p = BT + (size_t)col * ldb + kb;
  union { v4u u[2]; v16bf v; } u;
  u.u[0] = *(const v4u*)(p);
  u.u[1] = *(const v4u*)(p + 8);
  return u.v;
}

__device__ __forceinline__ float sigmoidf_(float x) {
  return 1.0f / (1.0f + __expf(-x));
}

// ---------------------------------------------------------------------------
// LDS-tiled transpose + fp32->bf16 convert:  W[K][N] -> WT[N][K]
// ---------------------------------------------------------------------------
__global__ __launch_bounds__(256) void transpose_cvt_k(
    const float* __restrict__ W, __bf16* __restrict__ WT, int K, int N) {
  __shared__ __bf16 tile[32][33];
  int ntiles = N >> 5;
  int bx = blockIdx.x % ntiles;   // N tile
  int by = blockIdx.x / ntiles;   // K tile
  int tx = threadIdx.x & 31;
  int ty = threadIdx.x >> 5;      // 0..7
#pragma unroll
  for (int i = 0; i < 32; i += 8) {
    int k = (by << 5) + ty + i;
    int n = (bx << 5) + tx;
    tile[ty + i][tx] = (__bf16)W[(size_t)k * N + n];
  }
  __syncthreads();
#pragma unroll
  for (int i = 0; i < 32; i += 8) {
    int n = (bx << 5) + ty + i;
    int k = (by << 5) + tx;
    WT[(size_t)n * K + k] = tile[tx][ty + i];
  }
}

// ---------------------------------------------------------------------------
// Embedding gather -> bf16 rows  (one block per (b,t) row)
// ---------------------------------------------------------------------------
__global__ __launch_bounds__(256) void embed_k(
    const int* __restrict__ x, const float* __restrict__ emb,
    __bf16* __restrict__ e) {
  int row = blockIdx.x;                 // b*T + t
  int idx = x[row];
  const float* src = emb + (size_t)idx * HH;
  __bf16* dst = e + (size_t)row * HH;
  for (int i = threadIdx.x; i < HH; i += blockDim.x)
    dst[i] = (__bf16)src[i];
}

// ---------------------------------------------------------------------------
// Init recurrent state: h0 -> bf16 h buffer, c0 -> fp32 c buffer
// ---------------------------------------------------------------------------
__global__ __launch_bounds__(256) void init_state_k(
    const float* __restrict__ h0, const float* __restrict__ c0,
    __bf16* __restrict__ hbuf, float* __restrict__ cbuf) {
  int i = blockIdx.x * blockDim.x + threadIdx.x;   // over B*H
  hbuf[i] = (__bf16)h0[i];
  cbuf[i] = c0[i];
}

// ---------------------------------------------------------------------------
// Input projection GEMM:  C[M][4H] = A[M][H](bf16) * WxT^T + bias
//   one wave -> 32(M) x 64(N) tile: 8 accumulators. Explicit ping-pong
//   double buffer: loads land directly in the slot the next WMMA consumes.
// ---------------------------------------------------------------------------
__global__ __launch_bounds__(256) void gemm_xz_k(
    const __bf16* __restrict__ A, const __bf16* __restrict__ BT,
    const float* __restrict__ bias, float* __restrict__ C, int M) {
  int wave = blockIdx.x * (blockDim.x >> 5) + (threadIdx.x >> 5);
  const int NT = G4H / 64;              // 64 column tiles
  int mt = wave / NT;
  int nt = wave % NT;
  int m0 = mt << 5;                     // 32 rows per wave
  int n0 = nt << 6;
  if (m0 >= M) return;

  v8f acc0[4] = {};
  v8f acc1[4] = {};
  const __bf16* arow0 = A + (size_t)m0 * HH;
  const __bf16* arow1 = arow0 + (size_t)16 * HH;

  v16bf a0[2], a1[2], bfr[2][4];

  // prologue: chunk 0 -> slot 0
  a0[0] = load_a_frag(arow0, HH);
  a1[0] = load_a_frag(arow1, HH);
#pragma unroll
  for (int j = 0; j < 4; ++j)
    bfr[0][j] = load_b_frag(BT + (size_t)(n0 + (j << 4)) * HH, HH);

  // 32 K-chunks: 15 double-iterations cover chunks 0..29
#pragma unroll 1
  for (int c = 0; c < 15; ++c) {
    int k1 = (2 * c + 1) << 5;
    int k2 = (2 * c + 2) << 5;
    a0[1] = load_a_frag(arow0 + k1, HH);         // slot1 <- chunk 2c+1
    a1[1] = load_a_frag(arow1 + k1, HH);
#pragma unroll
    for (int j = 0; j < 4; ++j)
      bfr[1][j] = load_b_frag(BT + (size_t)(n0 + (j << 4)) * HH + k1, HH);
#pragma unroll
    for (int j = 0; j < 4; ++j) {                // compute chunk 2c
      acc0[j] = wmma_bf16(a0[0], bfr[0][j], acc0[j]);
      acc1[j] = wmma_bf16(a1[0], bfr[0][j], acc1[j]);
    }
    a0[0] = load_a_frag(arow0 + k2, HH);         // slot0 <- chunk 2c+2
    a1[0] = load_a_frag(arow1 + k2, HH);
#pragma unroll
    for (int j = 0; j < 4; ++j)
      bfr[0][j] = load_b_frag(BT + (size_t)(n0 + (j << 4)) * HH + k2, HH);
#pragma unroll
    for (int j = 0; j < 4; ++j) {                // compute chunk 2c+1
      acc0[j] = wmma_bf16(a0[1], bfr[1][j], acc0[j]);
      acc1[j] = wmma_bf16(a1[1], bfr[1][j], acc1[j]);
    }
  }
  // epilogue: slot0 = chunk 30; load chunk 31 -> slot1; compute both
  {
    const int k31 = 31 << 5;
    a0[1] = load_a_frag(arow0 + k31, HH);
    a1[1] = load_a_frag(arow1 + k31, HH);
#pragma unroll
    for (int j = 0; j < 4; ++j)
      bfr[1][j] = load_b_frag(BT + (size_t)(n0 + (j << 4)) * HH + k31, HH);
#pragma unroll
    for (int j = 0; j < 4; ++j) {
      acc0[j] = wmma_bf16(a0[0], bfr[0][j], acc0[j]);
      acc1[j] = wmma_bf16(a1[0], bfr[0][j], acc1[j]);
    }
#pragma unroll
    for (int j = 0; j < 4; ++j) {
      acc0[j] = wmma_bf16(a0[1], bfr[1][j], acc0[j]);
      acc1[j] = wmma_bf16(a1[1], bfr[1][j], acc1[j]);
    }
  }

  int lane = threadIdx.x & 31;
  int col  = lane & 15;
  int mo   = (lane >> 4) << 3;
#pragma unroll
  for (int j = 0; j < 4; ++j) {
    int n = n0 + (j << 4) + col;
    float bv = bias[n];
#pragma unroll
    for (int r = 0; r < 8; ++r) {
      C[(size_t)(m0 + r + mo) * G4H + n]      = acc0[j][r] + bv;
      C[(size_t)(m0 + 16 + r + mo) * G4H + n] = acc1[j][r] + bv;
    }
  }
}

// ---------------------------------------------------------------------------
// Persistent fused LSTM scan (one launch per layer, loops over all T steps).
//   64 blocks (one per hidden tile) x 256 threads (8 waves = 8 batch tiles).
//   Wh slice (128 KB) staged into LDS once; c state in registers; xz loads
//   hoisted; A fragments ping-pong double-buffered; atomic grid barrier.
// ---------------------------------------------------------------------------
#define LDS_STRIDE 1032   // 1024 + 8 halves padding

__device__ __forceinline__ v16bf lds_b_frag(const __bf16* p) {
  union { v4u u[2]; v16bf v; } u;
  u.u[0] = *(const v4u*)(p);      // ds_load_b128
  u.u[1] = *(const v4u*)(p + 8);  // ds_load_b128
  return u.v;
}

__global__ __launch_bounds__(256) void lstm_scan_k(
    const __bf16* __restrict__ WhT, const float* __restrict__ xz,
    float* __restrict__ cbuf, __bf16* __restrict__ hA, __bf16* __restrict__ hB,
    __bf16* __restrict__ out_bf, float* __restrict__ out_f32,
    unsigned int* __restrict__ ctr, int s_base, int nsteps) {
  extern __shared__ __align__(16) char smem_raw[];
  __bf16* smem = (__bf16*)smem_raw;

  int ht = blockIdx.x;            // hidden tile 0..63
  int h0 = ht << 4;
  int w  = threadIdx.x >> 5;      // wave = batch tile 0..7
  int m0 = w << 4;
  int lane = threadIdx.x & 31;
  int col  = lane & 15;
  int mo   = (lane >> 4) << 3;
  int kb   = (lane >> 4) << 4;

  // ---- stage Wh slice into LDS (64 rows x 1024 halves, 16B chunks) -------
  for (int chunk = threadIdx.x; chunk < 64 * 128; chunk += blockDim.x) {
    int row = chunk >> 7;               // g*16 + c
    int off = (chunk & 127) << 3;       // halves
    int g = row >> 4, c = row & 15;
    const __bf16* src = WhT + (size_t)(g * HH + h0 + c) * HH + off;
    *(uint4*)(smem + (size_t)row * LDS_STRIDE + off) = *(const uint4*)src;
  }
  __syncthreads();

  // ---- c state in registers for the whole scan ---------------------------
  float creg[8];
#pragma unroll
  for (int r = 0; r < 8; ++r)
    creg[r] = cbuf[(size_t)(m0 + r + mo) * HH + h0 + col];

  for (int t = 0; t < nsteps; ++t) {
    int s = s_base + t;
    const __bf16* hprev = (s & 1) ? hB : hA;
    __bf16*       hnext = (s & 1) ? hA : hB;

    // ---- hoist xz gate loads: latency hidden under the recurrent GEMM ----
    float z[4][8];
#pragma unroll
    for (int r = 0; r < 8; ++r) {
      size_t zrow = ((size_t)(m0 + r + mo) * TT + t) * G4H + h0 + col;
#pragma unroll
      for (int g = 0; g < 4; ++g)
        z[g][r] = xz[zrow + g * HH];
    }

    // ---- z_tile += h_{t-1} @ Wh : ping-pong A, LDS-resident B ------------
    v8f acc[4] = {};
    const __bf16* arow = hprev + (size_t)m0 * HH;
    v16bf a[2];
    a[0] = load_a_frag(arow, HH);

#pragma unroll 1
    for (int c = 0; c < 15; ++c) {
      int k0 = (2 * c) << 5;
      int k1 = (2 * c + 1) << 5;
      int k2 = (2 * c + 2) << 5;
      a[1] = load_a_frag(arow + k1, HH);       // chunk 2c+1 in flight
#pragma unroll
      for (int g = 0; g < 4; ++g)              // compute chunk 2c
        acc[g] = wmma_bf16(a[0],
            lds_b_frag(smem + (size_t)((g << 4) + col) * LDS_STRIDE + k0 + kb),
            acc[g]);
      a[0] = load_a_frag(arow + k2, HH);       // chunk 2c+2 in flight
#pragma unroll
      for (int g = 0; g < 4; ++g)              // compute chunk 2c+1
        acc[g] = wmma_bf16(a[1],
            lds_b_frag(smem + (size_t)((g << 4) + col) * LDS_STRIDE + k1 + kb),
            acc[g]);
    }
    {
      const int k30 = 30 << 5, k31 = 31 << 5;
      a[1] = load_a_frag(arow + k31, HH);
#pragma unroll
      for (int g = 0; g < 4; ++g)              // compute chunk 30
        acc[g] = wmma_bf16(a[0],
            lds_b_frag(smem + (size_t)((g << 4) + col) * LDS_STRIDE + k30 + kb),
            acc[g]);
#pragma unroll
      for (int g = 0; g < 4; ++g)              // compute chunk 31
        acc[g] = wmma_bf16(a[1],
            lds_b_frag(smem + (size_t)((g << 4) + col) * LDS_STRIDE + k31 + kb),
            acc[g]);
    }

    // prefetch next timestep's xz gate lines
    if (t + 1 < nsteps) {
      size_t zn = ((size_t)(m0 + mo) * TT + (t + 1)) * G4H;
#pragma unroll
      for (int g = 0; g < 4; ++g)
        __builtin_prefetch(xz + zn + g * HH + h0 + col, 0, 3);
    }

    // ---- gates + cell update, fully in-register --------------------------
#pragma unroll
    for (int r = 0; r < 8; ++r) {
      int m = m0 + r + mo;
      float iv = sigmoidf_(acc[0][r] + z[0][r]);
      float fv = sigmoidf_(acc[1][r] + z[1][r]);
      float gv = tanhf(acc[2][r] + z[2][r]);
      float ov = sigmoidf_(acc[3][r] + z[3][r]);
      float cn = fv * creg[r] + iv * gv;
      creg[r] = cn;
      float hn = ov * tanhf(cn);
      hnext[(size_t)m * HH + h0 + col] = (__bf16)hn;
      size_t oidx = ((size_t)m * TT + t) * HH + h0 + col;
      if (out_bf)  out_bf[oidx]  = (__bf16)hn;
      if (out_f32) out_f32[oidx] = hn;
    }

    // ---- grid barrier between steps (monotonic counter) ------------------
    if (t + 1 < nsteps) {
      __threadfence();                  // make h stores device-visible
      __syncthreads();
      if (threadIdx.x == 0) {
        unsigned target = (unsigned)(s + 1) * NBLK;
        __hip_atomic_fetch_add(ctr, 1u, __ATOMIC_RELEASE,
                               __HIP_MEMORY_SCOPE_AGENT);
        while (__hip_atomic_load(ctr, __ATOMIC_ACQUIRE,
                                 __HIP_MEMORY_SCOPE_AGENT) < target)
          __builtin_amdgcn_s_sleep(1);
      }
      __syncthreads();
      __threadfence();                  // discard stale h lines
    }
  }

  // write back cell state (consumed by next layer / finalize)
#pragma unroll
  for (int r = 0; r < 8; ++r)
    cbuf[(size_t)(m0 + r + mo) * HH + h0 + col] = creg[r];
}

// ---------------------------------------------------------------------------
// Finalize: h2 = out2[:, T-1, :], c2 = cbuf
// ---------------------------------------------------------------------------
__global__ __launch_bounds__(256) void finalize_k(
    const float* __restrict__ out2, const float* __restrict__ cbuf,
    float* __restrict__ h2, float* __restrict__ c2) {
  int i = blockIdx.x * blockDim.x + threadIdx.x;   // over B*H
  int b = i >> 10;
  int h = i & (HH - 1);
  h2[i] = out2[((size_t)b * TT + (TT - 1)) * HH + h];
  c2[i] = cbuf[i];
}

// ---------------------------------------------------------------------------
// Host launcher
// ---------------------------------------------------------------------------
extern "C" void kernel_launch(void* const* d_in, const int* in_sizes, int n_in,
                              void* d_out, int out_size, void* d_ws, size_t ws_size,
                              hipStream_t stream) {
  (void)in_sizes; (void)n_in; (void)out_size; (void)ws_size;

  const int*   x    = (const int*)  d_in[0];
  const float* h0   = (const float*)d_in[1];
  const float* c0   = (const float*)d_in[2];
  const float* emb  = (const float*)d_in[3];
  const float* Wx   = (const float*)d_in[4];
  const float* Wh   = (const float*)d_in[5];
  const float* bias = (const float*)d_in[6];
  float* out = (float*)d_out;

  // Workspace carve-up (256B aligned slices)
  char* ws = (char*)d_ws;
  size_t off = 0;
  auto take = [&](size_t bytes) -> char* {
    char* p = ws + off;
    off = (off + bytes + 255) & ~(size_t)255;
    return p;
  };
  __bf16* WxT  = (__bf16*)take((size_t)G4H * HH * sizeof(__bf16));      //   8 MB
  __bf16* WhT  = (__bf16*)take((size_t)G4H * HH * sizeof(__bf16));      //   8 MB
  __bf16* e    = (__bf16*)take((size_t)BB * TT * HH * sizeof(__bf16));  //  64 MB
  __bf16* out1 = (__bf16*)take((size_t)BB * TT * HH * sizeof(__bf16));  //  64 MB
  float*  xz   = (float*) take((size_t)BB * TT * G4H * sizeof(float));  // 512 MB
  __bf16* hA   = (__bf16*)take((size_t)BB * HH * sizeof(__bf16));
  __bf16* hB   = (__bf16*)take((size_t)BB * HH * sizeof(__bf16));
  float*  cbuf = (float*) take((size_t)BB * HH * sizeof(float));
  unsigned int* ctr = (unsigned int*)take(256);

  // 0) reset grid-barrier counter (graph-capture-legal memset node)
  hipMemsetAsync(ctr, 0, sizeof(unsigned int), stream);

  // 1) Transpose + convert weights to bf16 [4H][H]
  int tr_blocks = (G4H / 32) * (HH / 32);             // 4096
  transpose_cvt_k<<<tr_blocks, 256, 0, stream>>>(Wx, WxT, HH, G4H);
  transpose_cvt_k<<<tr_blocks, 256, 0, stream>>>(Wh, WhT, HH, G4H);

  // 2) Embedding gather -> bf16
  embed_k<<<BB * TT, 256, 0, stream>>>(x, emb, e);

  // 3) Init recurrent state
  init_state_k<<<(BB * HH) / 256, 256, 0, stream>>>(h0, c0, hA, cbuf);

  const int gemm_blocks = ((BB * TT / 32) * (G4H / 64)) / 8;   // 8192
  const size_t scan_lds = (size_t)64 * LDS_STRIDE * sizeof(__bf16); // ~129 KB

  // 4) Layer 1: xz = e @ Wx + b, then persistent 256-step WMMA scan
  gemm_xz_k<<<gemm_blocks, 256, 0, stream>>>(e, WxT, bias, xz, BB * TT);
  lstm_scan_k<<<NBLK, 256, scan_lds, stream>>>(
      WhT, xz, cbuf, hA, hB, out1, (float*)nullptr, ctr, 0, TT);

  // 5) Layer 2 (same weights, state carries over): xz = out1 @ Wx + b
  gemm_xz_k<<<gemm_blocks, 256, 0, stream>>>(out1, WxT, bias, xz, BB * TT);
  lstm_scan_k<<<NBLK, 256, scan_lds, stream>>>(
      WhT, xz, cbuf, hA, hB, (__bf16*)nullptr, out, ctr, TT, TT);

  // 6) h2 / c2 tail of d_out
  float* h2 = out + (size_t)BB * TT * HH;
  float* c2 = h2 + (size_t)BB * HH;
  finalize_k<<<(BB * HH) / 256, 256, 0, stream>>>(out, cbuf, h2, c2);
}
